// NIQE_48687749267622
// MI455X (gfx1250) — matrix operand order repaired and checked by reading, not compiled, not run on hardware
//
#include <hip/hip_runtime.h>

// ---------------------------------------------------------------------------
// NIQE-style fit: per-batch covariance S = X^T X via v_wmma_f32_16x16x32_bf16
// (f32 accumulate), split-K with atomic merge; then SPD solve + quadratic form.
// ---------------------------------------------------------------------------

typedef __attribute__((ext_vector_type(16))) __bf16   v16bf;
typedef __attribute__((ext_vector_type(2)))  __bf16   v2bf;
typedef __attribute__((ext_vector_type(8)))  float    v8f;
typedef __attribute__((ext_vector_type(4)))  float    f32x4;
typedef __attribute__((ext_vector_type(4)))  unsigned int u32x4;

#define D       96    // feature dim
#define DT      6     // 16x16 tiles per dim
#define KCHUNK  64    // rows staged per iteration (two K=32 WMMA slabs)
#define PITCHE  72    // bf16 elements per LDS row = 144 B = 9 x 16 B (bank-friendly)

// ---------------------------------------------------------------------------
// Kernel 0: zero workspace accumulators (split-K uses atomics; runs every call)
// ---------------------------------------------------------------------------
__global__ void niqe_zero_kernel(float* __restrict__ p, int n) {
    int i = blockIdx.x * 256 + threadIdx.x;
    if (i < n) p[i] = 0.0f;
}

// ---------------------------------------------------------------------------
// Kernel 1: S[b] += X_b^T X_b, colsum[b] += column sums.
// grid = (K-segments, batches), block = 256 (8 waves of 32).
// ---------------------------------------------------------------------------
__global__ __launch_bounds__(256)
void niqe_cov_kernel(const float* __restrict__ X, int N, long long batchStride,
                     float* __restrict__ S, float* __restrict__ colsum)
{
    __shared__ __attribute__((aligned(16))) __bf16 tile[D * PITCHE]; // 13824 B
    __shared__ float colacc[D];

    const int tid  = threadIdx.x;
    const int lane = tid & 31;
    const int wave = tid >> 5;
    const int b    = blockIdx.y;
    const int seg  = blockIdx.x;
    const int nseg = gridDim.x;
    const float* __restrict__ Xb = X + (long long)b * batchStride;

    if (tid < D) colacc[tid] = 0.0f;

    // Upper-triangle tile assignment: wave w owns tiles w, w+8, w+16 of 21.
    int nt = 0, TI[3] = {0,0,0}, TJ[3] = {0,0,0};
    {
        int cnt = 0;
        for (int i = 0; i < DT; ++i)
            for (int j = i; j < DT; ++j) {
                if ((cnt & 7) == wave && nt < 3) { TI[nt] = i; TJ[nt] = j; ++nt; }
                ++cnt;
            }
    }
    const int ntu = __builtin_amdgcn_readfirstlane(nt);  // scalar guard: no EXEC games

    v8f acc[3];
    #pragma unroll
    for (int t = 0; t < 3; ++t) acc[t] = (v8f){0.f,0.f,0.f,0.f,0.f,0.f,0.f,0.f};

    const int chunksTotal = (N + KCHUNK - 1) / KCHUNK;
    const int cpseg  = (chunksTotal + nseg - 1) / nseg;
    const int rowBeg = seg * cpseg * KCHUNK;
    const int rowEnd = min(rowBeg + cpseg * KCHUNK, N);

    // Per-thread staging items: 3 x (row-pair p, 4-col group g); 32x24 = 768 items.
    int ip[3], ig[3];
    #pragma unroll
    for (int e = 0; e < 3; ++e) {
        int item = e * 256 + tid;
        ip[e] = item / 24;
        ig[e] = item - 24 * ip[e];
    }

    f32x4 ra[3], rb[3];
    auto loadChunk = [&](int chunk) {
        const f32x4 z = {0.f, 0.f, 0.f, 0.f};
        #pragma unroll
        for (int e = 0; e < 3; ++e) {
            int r0 = chunk + 2 * ip[e], r1 = r0 + 1;
            ra[e] = (r0 < rowEnd) ? *(const f32x4*)(Xb + (long long)r0 * D + 4 * ig[e]) : z;
            rb[e] = (r1 < rowEnd) ? *(const f32x4*)(Xb + (long long)r1 * D + 4 * ig[e]) : z;
        }
    };

    __syncthreads();
    if (rowBeg < rowEnd) loadChunk(rowBeg);           // software pipeline preload

    for (int chunk = rowBeg; chunk < rowEnd; chunk += KCHUNK) {
        // ---- store stage: regs -> LDS bf16, transposed, row-pair packed b32 ----
        #pragma unroll
        for (int e = 0; e < 3; ++e) {
            #pragma unroll
            for (int j = 0; j < 4; ++j) {
                int c = 4 * ig[e] + j;
                v2bf pk;                               // v_cvt_pk_bf16_f32-friendly
                pk[0] = (__bf16)ra[e][j];
                pk[1] = (__bf16)rb[e][j];
                *(v2bf*)(tile + c * PITCHE + 2 * ip[e]) = pk;
                atomicAdd(&colacc[c], ra[e][j] + rb[e][j]);   // ds_add_f32 (zeros in tail OK)
            }
        }
        __syncthreads();

        // Issue next chunk's global loads now: latency hides under WMMA below.
        if (chunk + KCHUNK < rowEnd) loadChunk(chunk + KCHUNK);
        // Long-range prefetch of chunk+2 (24 KB = 192 x 128 B lines).
        if (chunk + 2 * KCHUNK < rowEnd && tid < 192)
            __builtin_prefetch((const char*)(Xb + (long long)(chunk + 2 * KCHUNK) * D) + tid * 128, 0, 0);

        // ---- compute: 2 K=32 slabs x up to 3 tiles, EXEC all-1s throughout ----
        const int half = lane >> 4;    // 0|1
        const int mn   = lane & 15;    // A row-in-tile / B col-in-tile
        #pragma unroll
        for (int t = 0; t < 3; ++t) {
            if (t >= ntu) break;       // scalar branch (s_cbranch), not exec mask
            const u32x4* abase = (const u32x4*)tile + (TI[t] * 16 + mn) * 9;
            const u32x4* bbase = (const u32x4*)tile + (TJ[t] * 16 + mn) * 9;
            #pragma unroll
            for (int s = 0; s < 2; ++s) {              // slab: K = [32s, 32s+32)
                union { u32x4 q[2]; v16bf v; } A, Bm;
                // A 16x32 bf16: lane m(+16h) holds K in [8h,8h+8) then [16+8h,+8)
                A.q[0]  = abase[4 * s + half];
                A.q[1]  = abase[4 * s + 2 + half];
                // B 32x16 bf16: lane n(+16h) holds K in [8h,8h+8) then [16+8h,+8), col = n
                Bm.q[0] = bbase[4 * s + half];
                Bm.q[1] = bbase[4 * s + 2 + half];
                acc[t] = __builtin_amdgcn_wmma_f32_16x16x32_bf16(
                           false, A.v, false, Bm.v, (short)0, acc[t], false, false);
            }
        }
        __syncthreads();
    }

    // ---- split-K merge: atomic add tiles (and symmetric mirror) into S[b] ----
    float* __restrict__ Sb = S + (long long)b * (D * D);
    const int half = lane >> 4;
    const int ncol = lane & 15;
    for (int t = 0; t < ntu; ++t) {
        const int i = TI[t], j = TJ[t];
        #pragma unroll
        for (int r = 0; r < 8; ++r) {
            // C/D layout: lane l, vgpr r -> M = r + 8*(l>>4), N = l&15
            int M  = i * 16 + r + 8 * half;
            int Nc = j * 16 + ncol;
            float val = acc[t][r];
            atomicAdd(&Sb[M * D + Nc], val);
            if (i != j) atomicAdd(&Sb[Nc * D + M], val);
        }
    }
    if (tid < D) atomicAdd(&colsum[b * D + tid], colacc[tid]);
}

// ---------------------------------------------------------------------------
// Kernel 2: per batch -- cov = 0.5*(sigma_r+sigma_t) + 1e-3*I in LDS,
// solve cov*y = (mu_r - mu_t) (SPD => pivot-free), out[b] = sqrt(d . y).
// ---------------------------------------------------------------------------
__global__ __launch_bounds__(256)
void niqe_finalize_kernel(const float* __restrict__ S_t, const float* __restrict__ sum_t,
                          const float* __restrict__ S_r, const float* __restrict__ sum_r,
                          float* __restrict__ out, int N, int Np)
{
    __shared__ float cov[D * (D + 1)];     // pitch 97 (bank-friendly)
    __shared__ float mu_t[D], mu_r[D], dvec[D], yvec[D];

    const int tid = threadIdx.x;
    const int b   = blockIdx.x;

    if (tid < D) {
        float mt = sum_t[b * D + tid] * (1.0f / (float)N);
        float mr = sum_r[tid]         * (1.0f / (float)Np);
        mu_t[tid] = mt;
        mu_r[tid] = mr;
        dvec[tid] = mr - mt;
        yvec[tid] = mr - mt;
    }
    __syncthreads();

    const float sT = 1.0f / (float)(N - 1);
    const float sR = 1.0f / (float)(Np - 1);
    for (int idx = tid; idx < D * D; idx += 256) {
        int dd = idx / D, ee = idx - dd * D;
        float st = (S_t[(long long)b * D * D + idx] - (float)N  * mu_t[dd] * mu_t[ee]) * sT;
        float sr = (S_r[idx]                        - (float)Np * mu_r[dd] * mu_r[ee]) * sR;
        cov[dd * (D + 1) + ee] = 0.5f * (st + sr) + ((dd == ee) ? 0.001f : 0.0f);
    }
    __syncthreads();

    // Forward elimination (SPD, no pivoting). 8 row-waves x 32 col-lanes.
    const int ty = tid >> 5, tx = tid & 31;
    for (int k = 0; k < D; ++k) {
        float invp = 1.0f / cov[k * (D + 1) + k];
        for (int r = k + 1 + ty; r < D; r += 8) {
            float f = cov[r * (D + 1) + k] * invp;
            for (int c = k + 1 + tx; c < D; c += 32)
                cov[r * (D + 1) + c] -= f * cov[k * (D + 1) + c];
            if (tx == 0) yvec[r] -= f * yvec[k];
        }
        __syncthreads();
    }

    // Back substitution + quadratic form: wave 0 only (wave32).
    if (tid < 32) {
        for (int k = D - 1; k >= 0; --k) {
            float s = 0.0f;
            for (int c = k + 1 + tid; c < D; c += 32)
                s += cov[k * (D + 1) + c] * yvec[c];
            #pragma unroll
            for (int off = 16; off > 0; off >>= 1) s += __shfl_down(s, off, 32);
            if (tid == 0) yvec[k] = (yvec[k] - s) / cov[k * (D + 1) + k];
            __builtin_amdgcn_wave_barrier();   // keep same-wave LDS RAW ordered
        }
        float q = 0.0f;
        for (int c = tid; c < D; c += 32) q += dvec[c] * yvec[c];
        #pragma unroll
        for (int off = 16; off > 0; off >>= 1) q += __shfl_down(q, off, 32);
        if (tid == 0) out[b] = sqrtf(q);
    }
}

// ---------------------------------------------------------------------------
extern "C" void kernel_launch(void* const* d_in, const int* in_sizes, int n_in,
                              void* d_out, int out_size, void* d_ws, size_t ws_size,
                              hipStream_t stream)
{
    const float* x = (const float*)d_in[0];   // [B, N, 96] f32
    const float* p = (const float*)d_in[1];   // [Np, 96]   f32
    float* out = (float*)d_out;               // [B] f32

    const int B  = out_size;                  // 64
    const int N  = in_sizes[0] / (B * D);     // 10000
    const int Np = in_sizes[1] / D;           // 5000

    // Workspace layout (f32): S_t[B*D*D] | sum_t[B*D] | S_r[D*D] | sum_r[D]
    float* ws    = (float*)d_ws;
    float* S_t   = ws;
    float* sum_t = S_t   + (size_t)B * D * D;
    float* S_r   = sum_t + (size_t)B * D;
    float* sum_r = S_r   + (size_t)D * D;
    const int totalF = B * D * D + B * D + D * D + D;   // ~605K floats (2.4 MB)

    niqe_zero_kernel<<<(totalF + 255) / 256, 256, 0, stream>>>(ws, totalF);

    // x pass: 16 K-segments x 64 batches = 1024 workgroups
    dim3 gx(16, B);
    niqe_cov_kernel<<<gx, 256, 0, stream>>>(x, N, (long long)N * D, S_t, sum_t);

    // p pass: single "batch", 8 K-segments
    dim3 gp(8, 1);
    niqe_cov_kernel<<<gp, 256, 0, stream>>>(p, Np, 0, S_r, sum_r);

    niqe_finalize_kernel<<<B, 256, 0, stream>>>(S_t, sum_t, S_r, sum_r, out, N, Np);
}
